// Mymodel_42949673577
// MI455X (gfx1250) — matrix operand order repaired and checked by reading, not compiled, not run on hardware
//
#include <hip/hip_runtime.h>
#include <hip/hip_bf16.h>
#include <math.h>

typedef __attribute__((ext_vector_type(16))) _Float16 v16h;
typedef __attribute__((ext_vector_type(8)))  _Float16 v8h;
typedef __attribute__((ext_vector_type(8)))  float    v8f;

#define S_LEN  512
#define BATCH  64
#define EMB    256
#define HID    128
#define GATES  512   // 4*HID
#define CAND   32000
#define TSTEPS 128

__device__ __forceinline__ float sigmoidf_(float x) { return 1.0f / (1.0f + __expf(-x)); }

// ---------------------------------------------------------------------------
// One 16x16 f32 tile of  A(MxK) @ W(NxK)^T  via v_wmma_f32_16x16x32_f16.
// A, W are row-major f16 (pre-cast); fragments are built from aligned 16-byte
// vector loads (2x v8h per operand per K-step) following CDNA5 ISA 7.12.2:
//   A 16x32 f16 : grp0 (lanes 0-15)  row m=lane,    K = {k0..+7, k0+16..+23}
//                 grp1 (lanes 16-31) row m=lane-16, K = {k0+8..+15, k0+24..+31}
//   B 32x16 f16 : lanes 0-15 K = {k0..+7, k0+16..+23}, N = lane&15
//                 lanes 16-31 K = {k0+8..+15, k0+24..+31}
//   C/D v8f     : M = m0 + grp*8 + r, N = n0 + (lane&15)
// ---------------------------------------------------------------------------
__device__ __forceinline__ v8f wmma_tile_h(const _Float16* __restrict__ A, int lda,
                                           const _Float16* __restrict__ W, int ldw,
                                           int m0, int n0, int K, v8f acc)
{
    const int lane = threadIdx.x & 31;
    const int grp  = lane >> 4;
    const int l16  = lane & 15;
    const _Float16* arow = A + (size_t)(m0 + l16) * lda + grp * 8;
    const _Float16* wrow = W + (size_t)(n0 + l16) * ldw + grp * 8;
    for (int k0 = 0; k0 < K; k0 += 32) {
        __builtin_prefetch(arow + k0 + 128, 0, 1);
        __builtin_prefetch(wrow + k0 + 128, 0, 1);
        const v8h a0 = *(const v8h*)(arow + k0);
        const v8h a1 = *(const v8h*)(arow + k0 + 16);
        const v8h b0 = *(const v8h*)(wrow + k0);
        const v8h b1 = *(const v8h*)(wrow + k0 + 16);
        const v16h a = __builtin_shufflevector(a0, a1, 0,1,2,3,4,5,6,7,8,9,10,11,12,13,14,15);
        const v16h b = __builtin_shufflevector(b0, b1, 0,1,2,3,4,5,6,7,8,9,10,11,12,13,14,15);
        acc = __builtin_amdgcn_wmma_f32_16x16x32_f16(false, a, false, b,
                                                     (short)0, acc, false, false);
    }
    return acc;
}

// ---------------------------------------------------------------------------
// f32 -> f16 cast (grid-stride)
// ---------------------------------------------------------------------------
__global__ void __launch_bounds__(256)
cast_f16_kernel(const float* __restrict__ src, _Float16* __restrict__ dst, int n)
{
    for (int i = blockIdx.x * blockDim.x + threadIdx.x; i < n; i += gridDim.x * blockDim.x)
        dst[i] = (_Float16)src[i];
}

// ---------------------------------------------------------------------------
// Generic tiled GEMM: O = A@W^T (+ A2@W2^T) (+ bias0 + bias1). One 16x16 tile
// per wave; OutT selects f32/f16 output. TRANSB64: store in per-64-row-block
// transposed layout O[(m/64)*N + n][m%64] so serial consumers can load the
// m-axis contiguously (one b128 f16 load instead of 8 strided u16 loads).
// ---------------------------------------------------------------------------
template <typename OutT, bool TRANSB64>
__global__ void __launch_bounds__(256)
gemm_awt_kernel(const _Float16* __restrict__ A, int lda,
                const _Float16* __restrict__ W, int ldw, int K,
                const _Float16* __restrict__ A2, int lda2,
                const _Float16* __restrict__ W2, int ldw2, int K2,
                const float* __restrict__ bias0, const float* __restrict__ bias1,
                OutT* __restrict__ O, int ldo, int tilesM, int tilesN)
{
    const int wave = (int)((blockIdx.x * blockDim.x + threadIdx.x) >> 5);
    const int ntiles = tilesM * tilesN;
    if (wave >= ntiles) return;                       // wave-uniform
    const int m0 = (wave / tilesN) * 16;
    const int n0 = (wave % tilesN) * 16;
    v8f acc = {};
    acc = wmma_tile_h(A, lda, W, ldw, m0, n0, K, acc);
    if (A2) acc = wmma_tile_h(A2, lda2, W2, ldw2, m0, n0, K2, acc);
    const int lane = threadIdx.x & 31, grp = lane >> 4, l16 = lane & 15;
    const int n = n0 + l16;
    const float bb = (bias0 ? bias0[n] : 0.f) + (bias1 ? bias1[n] : 0.f);
    const int N = tilesN * 16;
#pragma unroll
    for (int r = 0; r < 8; ++r) {
        const int m = m0 + grp * 8 + r;
        if (TRANSB64)
            O[((size_t)(m >> 6) * N + n) * 64 + (m & 63)] = (OutT)(acc[r] + bb);
        else
            O[(size_t)m * ldo + n] = (OutT)(acc[r] + bb);
    }
}

// ---------------------------------------------------------------------------
// Sequential bidirectional LSTM encoder: block 0 = forward, block 1 = backward.
// h state lives in LDS as f16 (WMMA A operand); per step only h@Whh^T is
// computed. gx = precomputed x@Wih^T + biases, stored f16 in [s][g][b] layout
// so the per-step epilogue load is a single contiguous v8h (b128) per lane.
// ---------------------------------------------------------------------------
__global__ void __launch_bounds__(1024)
encoder_scan_kernel(const _Float16* __restrict__ gx,     // [2][S][G][B]
                    const _Float16* __restrict__ Whh_fh,
                    const _Float16* __restrict__ Whh_bh,
                    float* __restrict__ h_s,             // [S][B][2H]
                    _Float16* __restrict__ h_s_h,        // [S][B][2H]
                    float* __restrict__ cstate,          // [2][B][H]
                    float* __restrict__ gtmp)            // [2][B][G]
{
    const int dir = blockIdx.x;
    const _Float16* gxd = gx + (size_t)dir * S_LEN * GATES * BATCH;
    const _Float16* Whh = (dir == 0) ? Whh_fh : Whh_bh;
    float* cst = cstate + (size_t)dir * BATCH * HID;
    float* gt  = gtmp   + (size_t)dir * BATCH * GATES;

    __shared__ __align__(16) _Float16 sh_h[BATCH * HID];   // 16 KB

    for (int i = threadIdx.x; i < BATCH * HID; i += blockDim.x) {
        sh_h[i] = (_Float16)0.f; cst[i] = 0.f;
    }
    __syncthreads();

    const int wave = threadIdx.x >> 5;
    const int lane = threadIdx.x & 31, grp = lane >> 4, l16 = lane & 15;

    for (int t = 0; t < S_LEN; ++t) {
        const int s = (dir == 0) ? t : (S_LEN - 1 - t);
        const _Float16* gxs = gxd + (size_t)s * GATES * BATCH;

        // gates = sh_h @ Whh^T + gx[s]    (4 x 32 tiles over 32 waves)
        for (int tile = wave; tile < (BATCH / 16) * (GATES / 16); tile += 32) {
            const int m0 = (tile / (GATES / 16)) * 16;
            const int n0 = (tile % (GATES / 16)) * 16;
            v8f acc = {};
            acc = wmma_tile_h(sh_h, HID, Whh, HID, m0, n0, HID, acc);
            const int n = n0 + l16;
            // gx for this lane: fixed gate n, batch rows m0+grp*8 .. +7 (contiguous)
            const v8h gxv = *(const v8h*)(gxs + (size_t)n * BATCH + m0 + grp * 8);
#pragma unroll
            for (int r = 0; r < 8; ++r) {
                const int b = m0 + grp * 8 + r;
                gt[b * GATES + n] = acc[r] + (float)gxv[r];
            }
        }
        __syncthreads();

        // elementwise LSTM cell (PyTorch gate order i,f,g,o)
        for (int i = threadIdx.x; i < BATCH * HID; i += blockDim.x) {
            const int b = i >> 7, h = i & 127;
            const float gi = gt[b * GATES + h];
            const float gf = gt[b * GATES + HID + h];
            const float gg = gt[b * GATES + 2 * HID + h];
            const float go = gt[b * GATES + 3 * HID + h];
            const float c  = sigmoidf_(gf) * cst[i] + sigmoidf_(gi) * tanhf(gg);
            const float hv = sigmoidf_(go) * tanhf(c);
            cst[i] = c;
            sh_h[i] = (_Float16)hv;
            const size_t o = ((size_t)s * BATCH + b) * (2 * HID) + dir * HID + h;
            h_s[o]   = hv;
            h_s_h[o] = (_Float16)hv;
        }
        __syncthreads();
    }
}

// ---------------------------------------------------------------------------
// Decoder: single block (1024 threads), T sequential steps. WMMA for s@v_a^T
// and the gate GEMM; VALU for the attention dot products / softmax / ctx.
// Decoder hidden s lives in LDS as f16 (WMMA A); ctx/sfin stored f16 (they
// are only ever consumed as WMMA A operands).
// ---------------------------------------------------------------------------
__global__ void __launch_bounds__(1024)
decoder_kernel(const float* __restrict__ h_s,      // [S][B][2H] f32
               const float* __restrict__ wa_hs,    // [S][B][H]  f32
               const float* __restrict__ smask,    // [S][B]
               const _Float16* __restrict__ v_ah,  // [H][H]
               const float* __restrict__ u_a,      // [H]
               const _Float16* __restrict__ dWihh, // [G][2H]
               const _Float16* __restrict__ dWhhh, // [G][H]
               const float* __restrict__ dbih,
               const float* __restrict__ dbhh,
               float* __restrict__ sv,             // [B][H]
               float* __restrict__ esc,            // [S][B]
               _Float16* __restrict__ ctx_h,       // [B][2H]
               float* __restrict__ gdec,           // [B][G]
               float* __restrict__ ycell,          // [B][H]
               _Float16* __restrict__ sfin_h)      // [B][H]
{
    __shared__ __align__(16) _Float16 sh_s[BATCH * HID];   // 16 KB
    __shared__ float sh_red[1024];
    __shared__ float sh_max[BATCH];
    __shared__ float sh_inv[BATCH];

    for (int i = threadIdx.x; i < BATCH * HID; i += blockDim.x) {
        sh_s[i] = (_Float16)0.f; ycell[i] = 0.f;
    }
    __syncthreads();

    const int wave = threadIdx.x >> 5;
    const int lane = threadIdx.x & 31, grp = lane >> 4, l16 = lane & 15;
    const int tid = threadIdx.x;

    for (int t = 0; t < TSTEPS; ++t) {
        // 1) sv = s @ v_a^T   (4x8 = 32 tiles, one per wave)
        {
            const int m0 = (wave / 8) * 16;
            const int n0 = (wave % 8) * 16;
            v8f acc = {};
            acc = wmma_tile_h(sh_s, HID, v_ah, HID, m0, n0, HID, acc);
            const int n = n0 + l16;
#pragma unroll
            for (int r = 0; r < 8; ++r)
                sv[(m0 + grp * 8 + r) * HID + n] = acc[r];
        }
        __syncthreads();

        // 2) e[s,b] = u_a . tanh(wa_hs[s,b,:] + sv[b,:]) * mask
        for (int idx = tid; idx < S_LEN * BATCH; idx += 1024) {
            const int b = idx & (BATCH - 1);
            const int s_ = idx >> 6;
            const float* wh  = wa_hs + ((size_t)s_ * BATCH + b) * HID;
            const float* svb = sv + b * HID;
            float acc = 0.f;
            for (int h = 0; h < HID; ++h) acc += u_a[h] * tanhf(wh[h] + svb[h]);
            esc[s_ * BATCH + b] = acc * smask[s_ * BATCH + b];
        }
        __syncthreads();

        // 3) softmax over s (per b): 64 groups x 16 threads
        {
            const int b = tid >> 4, sub = tid & 15;
            float mx = -3.4e38f;
            for (int s_ = sub; s_ < S_LEN; s_ += 16)
                mx = fmaxf(mx, esc[s_ * BATCH + b]);
            sh_red[tid] = mx;
            __syncthreads();
            if (sub == 0) {
                float m = sh_red[tid];
#pragma unroll
                for (int i = 1; i < 16; ++i) m = fmaxf(m, sh_red[tid + i]);
                sh_max[b] = m;
            }
            __syncthreads();
            float sm = 0.f;
            for (int s_ = sub; s_ < S_LEN; s_ += 16) {
                const float v = __expf(esc[s_ * BATCH + b] - sh_max[b]);
                esc[s_ * BATCH + b] = v;
                sm += v;
            }
            sh_red[tid] = sm;
            __syncthreads();
            if (sub == 0) {
                float m = 0.f;
#pragma unroll
                for (int i = 0; i < 16; ++i) m += sh_red[tid + i];
                sh_inv[b] = 1.0f / m;
            }
            __syncthreads();
        }

        // ctx[b,e] = sum_s alpha * h_s   (f32 accumulation, stored f16)
        for (int idx = tid; idx < BATCH * 2 * HID; idx += 1024) {
            const int b = idx >> 8;
            const int e = idx & 255;
            float acc = 0.f;
            for (int s_ = 0; s_ < S_LEN; ++s_)
                acc += esc[s_ * BATCH + b] * h_s[((size_t)s_ * BATCH + b) * (2 * HID) + e];
            ctx_h[b * (2 * HID) + e] = (_Float16)(acc * sh_inv[b]);
        }
        __syncthreads();

        // 4) gates = ctx @ dWih^T + s @ dWhh^T + biases  (4x32 tiles)
        for (int tile = wave; tile < (BATCH / 16) * (GATES / 16); tile += 32) {
            const int m0 = (tile / (GATES / 16)) * 16;
            const int n0 = (tile % (GATES / 16)) * 16;
            v8f acc = {};
            acc = wmma_tile_h(ctx_h, 2 * HID, dWihh, 2 * HID, m0, n0, 2 * HID, acc);
            acc = wmma_tile_h(sh_s, HID, dWhhh, HID, m0, n0, HID, acc);
            const int n = n0 + l16;
            const float bb = dbih[n] + dbhh[n];
#pragma unroll
            for (int r = 0; r < 8; ++r)
                gdec[(m0 + grp * 8 + r) * GATES + n] = acc[r] + bb;
        }
        __syncthreads();

        // 5) LSTMCell update
        for (int i = tid; i < BATCH * HID; i += 1024) {
            const int b = i >> 7, h = i & 127;
            const float gi = gdec[b * GATES + h];
            const float gf = gdec[b * GATES + HID + h];
            const float gg = gdec[b * GATES + 2 * HID + h];
            const float go = gdec[b * GATES + 3 * HID + h];
            const float y  = sigmoidf_(gf) * ycell[i] + sigmoidf_(gi) * tanhf(gg);
            const float s  = sigmoidf_(go) * tanhf(y);
            ycell[i] = y;
            sh_s[i] = (_Float16)s;
            sfin_h[i] = (_Float16)s;
        }
        __syncthreads();
    }
}

// ---------------------------------------------------------------------------
// Row softmax over C=32000, one block per batch row.
// ---------------------------------------------------------------------------
__global__ void __launch_bounds__(256)
softmax_rows_kernel(const float* __restrict__ logits, float* __restrict__ out)
{
    const int b = blockIdx.x;
    const float* row = logits + (size_t)b * CAND;
    float* orow = out + (size_t)b * CAND;
    __shared__ float red[256];

    float mx = -3.4e38f;
    for (int c = threadIdx.x; c < CAND; c += 256) mx = fmaxf(mx, row[c]);
    red[threadIdx.x] = mx;
    __syncthreads();
    for (int st = 128; st > 0; st >>= 1) {
        if (threadIdx.x < st) red[threadIdx.x] = fmaxf(red[threadIdx.x], red[threadIdx.x + st]);
        __syncthreads();
    }
    mx = red[0];
    __syncthreads();

    float sm = 0.f;
    for (int c = threadIdx.x; c < CAND; c += 256) sm += __expf(row[c] - mx);
    red[threadIdx.x] = sm;
    __syncthreads();
    for (int st = 128; st > 0; st >>= 1) {
        if (threadIdx.x < st) red[threadIdx.x] += red[threadIdx.x + st];
        __syncthreads();
    }
    const float inv = 1.0f / red[0];
    for (int c = threadIdx.x; c < CAND; c += 256) orow[c] = __expf(row[c] - mx) * inv;
}

// ---------------------------------------------------------------------------
extern "C" void kernel_launch(void* const* d_in, const int* in_sizes, int n_in,
                              void* d_out, int out_size, void* d_ws, size_t ws_size,
                              hipStream_t stream)
{
    (void)in_sizes; (void)n_in; (void)out_size; (void)ws_size;
    const float* x     = (const float*)d_in[0];
    const float* smask = (const float*)d_in[1];
    const float* Wih_f = (const float*)d_in[2];
    const float* Whh_f = (const float*)d_in[3];
    const float* bih_f = (const float*)d_in[4];
    const float* bhh_f = (const float*)d_in[5];
    const float* Wih_b = (const float*)d_in[6];
    const float* Whh_b = (const float*)d_in[7];
    const float* bih_b = (const float*)d_in[8];
    const float* bhh_b = (const float*)d_in[9];
    const float* dWih  = (const float*)d_in[10];
    const float* dWhh  = (const float*)d_in[11];
    const float* dbih  = (const float*)d_in[12];
    const float* dbhh  = (const float*)d_in[13];
    const float* w_a   = (const float*)d_in[14];
    const float* v_a   = (const float*)d_in[15];
    const float* u_a   = (const float*)d_in[16];
    const float* w_b   = (const float*)d_in[17];
    const float* v_b   = (const float*)d_in[18];
    float* out = (float*)d_out;

    // workspace carve-out (~185 MB)
    char* ws = (char*)d_ws;
    size_t off = 0;
    auto carve = [&](size_t bytes) -> void* {
        void* p = ws + off;
        off = (off + bytes + 255) & ~(size_t)255;
        return p;
    };
    const size_t SB = (size_t)S_LEN * BATCH;
    _Float16* gx     = (_Float16*)carve(2 * SB * GATES * sizeof(_Float16)); // 67 MB  [2][S][G][B]
    float*    h_s    = (float*)carve(SB * 2 * HID * sizeof(float));         // 33.6 MB
    _Float16* h_s_h  = (_Float16*)carve(SB * 2 * HID * sizeof(_Float16));   // 16.8 MB
    float*    wa_hs  = (float*)carve(SB * HID * sizeof(float));             // 16.8 MB
    float*    logits = (float*)carve((size_t)BATCH * CAND * sizeof(float)); // 8.2 MB
    float*    enc_c  = (float*)carve(2 * BATCH * HID * sizeof(float));
    float*    enc_gt = (float*)carve(2 * BATCH * GATES * sizeof(float));
    float*    sv     = (float*)carve((size_t)BATCH * HID * sizeof(float));
    float*    esc    = (float*)carve(SB * sizeof(float));
    _Float16* ctx_h  = (_Float16*)carve((size_t)BATCH * 2 * HID * sizeof(_Float16));
    float*    gdec   = (float*)carve((size_t)BATCH * GATES * sizeof(float));
    float*    ycell  = (float*)carve((size_t)BATCH * HID * sizeof(float));
    _Float16* sfin_h = (_Float16*)carve((size_t)BATCH * HID * sizeof(_Float16));
    // f16 copies of GEMM operands
    _Float16* x_h     = (_Float16*)carve(SB * EMB * sizeof(_Float16));      // 16.8 MB
    _Float16* Wih_fh  = (_Float16*)carve((size_t)GATES * EMB * sizeof(_Float16));
    _Float16* Wih_bh  = (_Float16*)carve((size_t)GATES * EMB * sizeof(_Float16));
    _Float16* Whh_fh  = (_Float16*)carve((size_t)GATES * HID * sizeof(_Float16));
    _Float16* Whh_bh  = (_Float16*)carve((size_t)GATES * HID * sizeof(_Float16));
    _Float16* w_ah    = (_Float16*)carve((size_t)HID * 2 * HID * sizeof(_Float16));
    _Float16* v_ah    = (_Float16*)carve((size_t)HID * HID * sizeof(_Float16));
    _Float16* dWihh   = (_Float16*)carve((size_t)GATES * 2 * HID * sizeof(_Float16));
    _Float16* dWhhh   = (_Float16*)carve((size_t)GATES * HID * sizeof(_Float16));
    _Float16* w_bh    = (_Float16*)carve((size_t)CAND * HID * sizeof(_Float16));   // 8.2 MB
    _Float16* v_bh    = (_Float16*)carve((size_t)CAND * 2 * HID * sizeof(_Float16)); // 16.4 MB

    // K0: pre-cast all GEMM operands to f16 (pure bandwidth, ~85 MB total)
    auto cast = [&](const float* s, _Float16* d, size_t n) {
        int blocks = (int)((n + 1023) / 1024);
        if (blocks > 4096) blocks = 4096;
        cast_f16_kernel<<<blocks, 256, 0, stream>>>(s, d, (int)n);
    };
    cast(x,     x_h,    SB * EMB);
    cast(Wih_f, Wih_fh, (size_t)GATES * EMB);
    cast(Wih_b, Wih_bh, (size_t)GATES * EMB);
    cast(Whh_f, Whh_fh, (size_t)GATES * HID);
    cast(Whh_b, Whh_bh, (size_t)GATES * HID);
    cast(w_a,   w_ah,   (size_t)HID * 2 * HID);
    cast(v_a,   v_ah,   (size_t)HID * HID);
    cast(dWih,  dWihh,  (size_t)GATES * 2 * HID);
    cast(dWhh,  dWhhh,  (size_t)GATES * HID);
    cast(w_b,   w_bh,   (size_t)CAND * HID);
    cast(v_b,   v_bh,   (size_t)CAND * 2 * HID);

    // K1: gx[d] = x @ Wih[d]^T + biases  (M=32768, N=512, K=256), stored
    // per-64-row-block transposed ([s][g][b]) for the serial scan's benefit.
    {
        const int tilesM = (int)(SB / 16), tilesN = GATES / 16;      // 2048 x 32
        const int blocks = tilesM * tilesN / 8;                      // 8 waves/block
        gemm_awt_kernel<_Float16, true><<<blocks, 256, 0, stream>>>(
            x_h, EMB, Wih_fh, EMB, EMB, nullptr, 0, nullptr, 0, 0,
            bih_f, bhh_f, gx, 0, tilesM, tilesN);
        gemm_awt_kernel<_Float16, true><<<blocks, 256, 0, stream>>>(
            x_h, EMB, Wih_bh, EMB, EMB, nullptr, 0, nullptr, 0, 0,
            bih_b, bhh_b, gx + SB * GATES, 0, tilesM, tilesN);
    }

    // K2: bidirectional sequential LSTM scan (one WGP-resident block per dir)
    encoder_scan_kernel<<<2, 1024, 0, stream>>>(gx, Whh_fh, Whh_bh,
                                                h_s, h_s_h, enc_c, enc_gt);

    // K3: wa_hs = h_s @ w_a^T   (M=32768, N=128, K=256)
    {
        const int tilesM = (int)(SB / 16), tilesN = HID / 16;        // 2048 x 8
        const int blocks = tilesM * tilesN / 8;
        gemm_awt_kernel<float, false><<<blocks, 256, 0, stream>>>(
            h_s_h, 2 * HID, w_ah, 2 * HID, 2 * HID, nullptr, 0, nullptr, 0, 0,
            nullptr, nullptr, wa_hs, HID, tilesM, tilesN);
    }

    // K4: decoder loop (T=128 sequential steps, single block)
    decoder_kernel<<<1, 1024, 0, stream>>>(h_s, wa_hs, smask, v_ah, u_a,
                                           dWihh, dWhhh, dbih, dbhh,
                                           sv, esc, ctx_h, gdec, ycell, sfin_h);

    // K5: logits = sfin @ w_b^T + ctx @ v_b^T   (M=64, N=32000, K=128+256)
    {
        const int tilesM = BATCH / 16, tilesN = CAND / 16;           // 4 x 2000
        const int blocks = tilesM * tilesN / 8;                      // 1000
        gemm_awt_kernel<float, false><<<blocks, 256, 0, stream>>>(
            sfin_h, HID, w_bh, HID, HID, ctx_h, 2 * HID, v_bh, 2 * HID, 2 * HID,
            nullptr, nullptr, logits, CAND, tilesM, tilesN);
    }

    // K6: row softmax -> d_out
    softmax_rows_kernel<<<BATCH, 256, 0, stream>>>(logits, out);
}
